// ParticleAttentionLayer_55972013802027
// MI455X (gfx1250) — matrix operand order repaired and checked by reading, not compiled, not run on hardware
//
#include <hip/hip_runtime.h>
#include <hip/hip_bf16.h>

// ---------------- problem constants (from reference) ----------------
#define NN       50000      // nodes (50000 = 3125 * 16, no row guards needed)
#define DD       256        // feature dim
#define HH       8          // heads
#define CC       32         // channels per head
#define NE       1600000    // edges (reference appends NN self-loops)
#define NE_TOT   (NE + NN)
#define NEG_SLOPE 0.2f
#define LN_EPS    1e-5f

typedef __attribute__((ext_vector_type(16))) __bf16 v16bf;
typedef __attribute__((ext_vector_type(8)))  __bf16 v8bf;
typedef __attribute__((ext_vector_type(8)))  float  v8f;

// monotonic float <-> uint mapping for atomicMax-based segment max
__device__ __forceinline__ unsigned enc_f(float f) {
    unsigned u = __float_as_uint(f);
    return (u & 0x80000000u) ? ~u : (u | 0x80000000u);
}
__device__ __forceinline__ float dec_f(unsigned u) {
    u = (u & 0x80000000u) ? (u & 0x7FFFFFFFu) : ~u;
    return __uint_as_float(u);
}

// ---------------- 0) deterministic workspace init ----------------
__global__ __launch_bounds__(256) void gat_init(float* __restrict__ acc,
                                                unsigned* __restrict__ mmax,
                                                float* __restrict__ denom) {
    int i = blockIdx.x * 256 + threadIdx.x;
    if (i < NN * DD) acc[i] = 0.0f;
    if (i < NN * HH) { mmax[i] = 0u; denom[i] = 0.0f; }
}

// ---------------- 0b) fp32 -> bf16 prep: xb = bf16(x), Wt = bf16(W)^T ----------
__global__ __launch_bounds__(256) void gat_prep_bf16(const float* __restrict__ x,
                                                     const float* __restrict__ Wm,
                                                     __bf16* __restrict__ xb,
                                                     __bf16* __restrict__ Wt) {
    int i = blockIdx.x * 256 + threadIdx.x;
    if (i < NN * DD) xb[i] = (__bf16)x[i];
    if (i < DD * DD) {
        int k = i >> 8, n = i & 255;             // W row-major [k][n]
        Wt[(size_t)n * DD + k] = (__bf16)Wm[i];  // transposed: [n][k] contiguous in k
    }
}

// ---------------- 1) h = x @ W via WMMA bf16, async-to-LDS staging -------------
// block = 4 waves = 16 rows x 64 cols; grid = (3125, 4) covers 50000 x 256.
// LDS: A strip 16x256 bf16 (8KB, shared) + per-wave B strip 16x256 bf16 (8KB).
__global__ __launch_bounds__(128) void gat_gemm_wmma(const __bf16* __restrict__ xb,
                                                     const __bf16* __restrict__ Wt,
                                                     float* __restrict__ h) {
    __shared__ __attribute__((aligned(16))) __bf16 smem[16 * DD + 4 * 16 * DD]; // 40KB

    const int lane = threadIdx.x & 31;
    const int wave = threadIdx.x >> 5;
    const int half = lane >> 4;                   // ISA fragment lane-half
    const int l    = lane & 15;
    const int rowBase = blockIdx.x * 16;
    const int colBase = (blockIdx.y * 4 + wave) * 16;

    __bf16* sA = smem;                            // [16][256] rows of x
    __bf16* sB = smem + 16 * DD + wave * 16 * DD; // [16][256] cols of W (transposed)
    const unsigned ldsA = (unsigned)(size_t)(void*)sA;
    const unsigned ldsB = (unsigned)(size_t)(void*)sB;

    // --- async stage A strip (wave 0 only): 8KB = 16 iters x 32 lanes x 16B ---
    if (wave == 0) {
        const char* ga = (const char*)(xb + (size_t)rowBase * DD);
        #pragma unroll
        for (int it = 0; it < 16; ++it) {
            const int off = (it * 32 + lane) * 16;
            const unsigned la = ldsA + off;
            const char* gp = ga + off;
            asm volatile("global_load_async_to_lds_b128 %0, %1, off"
                         :: "v"(la), "v"(gp) : "memory");
        }
    }
    // --- async stage this wave's B strip: Wt rows [colBase, colBase+16) ---
    {
        const char* gb = (const char*)(Wt + (size_t)colBase * DD);
        #pragma unroll
        for (int it = 0; it < 16; ++it) {
            const int off = (it * 32 + lane) * 16;
            const unsigned lb = ldsB + off;
            const char* gp = gb + off;
            asm volatile("global_load_async_to_lds_b128 %0, %1, off"
                         :: "v"(lb), "v"(gp) : "memory");
        }
    }
    asm volatile("s_wait_asynccnt 0x0" ::: "memory");
    __syncthreads();

    v8f c = {0.f, 0.f, 0.f, 0.f, 0.f, 0.f, 0.f, 0.f};

    const __bf16* ar = sA + l * DD;               // A: row M = l
    const __bf16* br = sB + l * DD;               // B: col N = l (K contiguous)

    #pragma unroll
    for (int k0 = 0; k0 < DD; k0 += 32) {
        // A 16x32 frag: half0 -> K = k0+{0..7, 16..23}; half1 -> {8..15, 24..31}
        const int ka = k0 + half * 8;
        v8bf a0 = *(const v8bf*)(ar + ka);
        v8bf a1 = *(const v8bf*)(ar + ka + 16);
        v16bf a = __builtin_shufflevector(a0, a1, 0, 1, 2, 3, 4, 5, 6, 7,
                                                  8, 9, 10, 11, 12, 13, 14, 15);
        // B 32x16 frag: same K order as A: half0 -> {0..7, 16..23}; half1 -> {8..15, 24..31}
        const int kb = k0 + half * 8;
        v8bf b0 = *(const v8bf*)(br + kb);
        v8bf b1 = *(const v8bf*)(br + kb + 16);
        v16bf b = __builtin_shufflevector(b0, b1, 0, 1, 2, 3, 4, 5, 6, 7,
                                                  8, 9, 10, 11, 12, 13, 14, 15);

        c = __builtin_amdgcn_wmma_f32_16x16x32_bf16(false, a, false, b,
                                                    (short)0, c, false, false);
    }

    // C/D layout: VGPR r -> M = half*8 + r, N = l   (no guards: 50000 % 16 == 0)
    float* hp = h + (size_t)(rowBase + half * 8) * DD + colBase + l;
    #pragma unroll
    for (int r = 0; r < 8; ++r) hp[(size_t)r * DD] = c[r];
}

// ---------------- 2) per-(node,head) attention coefficients ----------------
__global__ __launch_bounds__(256) void gat_attn_coef(const float* __restrict__ h,
                                                     const float* __restrict__ att_src,
                                                     const float* __restrict__ att_dst,
                                                     float* __restrict__ asrc,
                                                     float* __restrict__ adst) {
    int idx = blockIdx.x * 256 + threadIdx.x;     // n*H + head
    if (idx >= NN * HH) return;
    int n  = idx >> 3;
    int hd = idx & 7;
    const float* hp = h + (size_t)n * DD + hd * CC;
    const float* as = att_src + hd * CC;
    const float* ad = att_dst + hd * CC;
    float ss = 0.f, sd = 0.f;
    #pragma unroll
    for (int i = 0; i < CC; ++i) {
        float v = hp[i];
        ss += v * as[i];
        sd += v * ad[i];
    }
    asrc[idx] = ss;
    adst[idx] = sd;
}

__device__ __forceinline__ void edge_nodes(const int* __restrict__ ei, int e,
                                           int& s, int& d) {
    if (e < NE) { s = ei[e]; d = ei[NE + e]; }
    else        { s = d = e - NE; }               // appended self-loops
}

// ---------------- 3) edge pass 1: segment max (encoded atomicMax) --------------
__global__ __launch_bounds__(256) void gat_edge_max(const int* __restrict__ ei,
                                                    const float* __restrict__ asrc,
                                                    const float* __restrict__ adst,
                                                    unsigned* __restrict__ mmax) {
    int e = blockIdx.x * 256 + threadIdx.x;
    if (e >= NE_TOT) return;
    int s, d; edge_nodes(ei, e, s, d);
    const float4* ap = (const float4*)(asrc + (size_t)s * HH);
    const float4* dp = (const float4*)(adst + (size_t)d * HH);
    float4 a0 = ap[0], a1 = ap[1], b0 = dp[0], b1 = dp[1];
    float ev[8] = {a0.x + b0.x, a0.y + b0.y, a0.z + b0.z, a0.w + b0.w,
                   a1.x + b1.x, a1.y + b1.y, a1.z + b1.z, a1.w + b1.w};
    #pragma unroll
    for (int hd = 0; hd < HH; ++hd) {
        float t = ev[hd];
        t = t > 0.f ? t : NEG_SLOPE * t;
        atomicMax(mmax + (size_t)d * HH + hd, enc_f(t));
    }
}

// ---------------- 4) edge pass 2: segment sum of exp(e - m) --------------------
__global__ __launch_bounds__(256) void gat_edge_sum(const int* __restrict__ ei,
                                                    const float* __restrict__ asrc,
                                                    const float* __restrict__ adst,
                                                    const unsigned* __restrict__ mmax,
                                                    float* __restrict__ denom) {
    int e = blockIdx.x * 256 + threadIdx.x;
    if (e >= NE_TOT) return;
    int s, d; edge_nodes(ei, e, s, d);
    const float4* ap = (const float4*)(asrc + (size_t)s * HH);
    const float4* dp = (const float4*)(adst + (size_t)d * HH);
    float4 a0 = ap[0], a1 = ap[1], b0 = dp[0], b1 = dp[1];
    float ev[8] = {a0.x + b0.x, a0.y + b0.y, a0.z + b0.z, a0.w + b0.w,
                   a1.x + b1.x, a1.y + b1.y, a1.z + b1.z, a1.w + b1.w};
    #pragma unroll
    for (int hd = 0; hd < HH; ++hd) {
        float t = ev[hd];
        t = t > 0.f ? t : NEG_SLOPE * t;
        float mm = dec_f(mmax[(size_t)d * HH + hd]);
        atomicAdd(denom + (size_t)d * HH + hd, __expf(t - mm));
    }
}

// ---------------- 5) edge pass 3: alpha-weighted scatter-add (wave per edge) ---
__global__ __launch_bounds__(256) void gat_edge_agg(const int* __restrict__ ei,
                                                    const float* __restrict__ asrc,
                                                    const float* __restrict__ adst,
                                                    const unsigned* __restrict__ mmax,
                                                    const float* __restrict__ denom,
                                                    const float* __restrict__ h,
                                                    float* __restrict__ acc) {
    int e = blockIdx.x * 8 + (threadIdx.x >> 5);
    if (e >= NE_TOT) return;
    const int lane = threadIdx.x & 31;             // lane covers cols [8*lane, 8*lane+8)
    int s, d; edge_nodes(ei, e, s, d);
    const int hd = lane >> 2;                      // head of this lane's 8 columns
    float t = asrc[(size_t)s * HH + hd] + adst[(size_t)d * HH + hd];
    t = t > 0.f ? t : NEG_SLOPE * t;
    float mm    = dec_f(mmax[(size_t)d * HH + hd]);
    float alpha = __expf(t - mm) / denom[(size_t)d * HH + hd];

    const float4* hp = (const float4*)(h + (size_t)s * DD + lane * 8);
    float4 v0 = hp[0], v1 = hp[1];
    float* ap = acc + (size_t)d * DD + lane * 8;
    atomicAdd(ap + 0, v0.x * alpha);
    atomicAdd(ap + 1, v0.y * alpha);
    atomicAdd(ap + 2, v0.z * alpha);
    atomicAdd(ap + 3, v0.w * alpha);
    atomicAdd(ap + 4, v1.x * alpha);
    atomicAdd(ap + 5, v1.y * alpha);
    atomicAdd(ap + 6, v1.z * alpha);
    atomicAdd(ap + 7, v1.w * alpha);
}

// ---------------- 6) bias + residual + LayerNorm (wave per node) ---------------
__global__ __launch_bounds__(256) void gat_final(const float* __restrict__ acc,
                                                 const float* __restrict__ x,
                                                 const float* __restrict__ bias,
                                                 const float* __restrict__ lnw,
                                                 const float* __restrict__ lnb,
                                                 float* __restrict__ out) {
    int n = blockIdx.x * 8 + (threadIdx.x >> 5);
    if (n >= NN) return;
    const int lane = threadIdx.x & 31;
    const float* ac = acc  + (size_t)n * DD + lane * 8;
    const float* xp = x    + (size_t)n * DD + lane * 8;
    const float* bp = bias + lane * 8;
    float y[8];
    float s = 0.f, s2 = 0.f;
    #pragma unroll
    for (int i = 0; i < 8; ++i) {
        y[i] = ac[i] + bp[i] + xp[i];
        s  += y[i];
        s2 += y[i] * y[i];
    }
    #pragma unroll
    for (int o = 16; o > 0; o >>= 1) {            // wave32 reduction
        s  += __shfl_xor(s,  o, 32);
        s2 += __shfl_xor(s2, o, 32);
    }
    float mu  = s * (1.0f / DD);
    float var = s2 * (1.0f / DD) - mu * mu;
    float inv = rsqrtf(var + LN_EPS);
    float* op = out + (size_t)n * DD + lane * 8;
    #pragma unroll
    for (int i = 0; i < 8; ++i)
        op[i] = (y[i] - mu) * inv * lnw[lane * 8 + i] + lnb[lane * 8 + i];
}

// ---------------- launch ----------------
extern "C" void kernel_launch(void* const* d_in, const int* in_sizes, int n_in,
                              void* d_out, int out_size, void* d_ws, size_t ws_size,
                              hipStream_t stream) {
    const float* x       = (const float*)d_in[0];
    const int*   ei      = (const int*)  d_in[1];
    const float* Wm      = (const float*)d_in[2];
    const float* att_src = (const float*)d_in[3];
    const float* att_dst = (const float*)d_in[4];
    const float* bias    = (const float*)d_in[5];
    const float* lnw     = (const float*)d_in[6];
    const float* lnb     = (const float*)d_in[7];
    float* out = (float*)d_out;

    // ws layout (floats): h[N*D] | acc[N*D] | asrc[N*H] | adst[N*H] | m[N*H] |
    //                     denom[N*H] | xb[N*D bf16] | Wt[D*D bf16]
    float*    h     = (float*)d_ws;
    float*    acc   = h    + (size_t)NN * DD;
    float*    asrc  = acc  + (size_t)NN * DD;
    float*    adst  = asrc + (size_t)NN * HH;
    unsigned* mmax  = (unsigned*)(adst + (size_t)NN * HH);
    float*    denom = (float*)(mmax + (size_t)NN * HH);
    __bf16*   xb    = (__bf16*)(denom + (size_t)NN * HH);
    __bf16*   Wt    = xb + (size_t)NN * DD;

    // 0) re-init accumulators every call (graph-replay deterministic)
    gat_init<<<(NN * DD + 255) / 256, 256, 0, stream>>>(acc, mmax, denom);
    // 0b) bf16 conversion + W transpose
    gat_prep_bf16<<<(NN * DD + 255) / 256, 256, 0, stream>>>(x, Wm, xb, Wt);

    // 1) h = x @ W  (WMMA bf16, LDS-staged via async loads)
    dim3 ggrid(NN / 16, 4);
    gat_gemm_wmma<<<ggrid, 128, 0, stream>>>(xb, Wt, h);

    // 2) attention coefficients
    gat_attn_coef<<<(NN * HH + 255) / 256, 256, 0, stream>>>(h, att_src, att_dst, asrc, adst);

    // 3-5) edge passes
    int eblocks = (NE_TOT + 255) / 256;
    gat_edge_max<<<eblocks, 256, 0, stream>>>(ei, asrc, adst, mmax);
    gat_edge_sum<<<eblocks, 256, 0, stream>>>(ei, asrc, adst, mmax, denom);
    gat_edge_agg<<<(NE_TOT + 7) / 8, 256, 0, stream>>>(ei, asrc, adst, mmax, denom, h, acc);

    // 6) epilogue: bias + residual + LayerNorm
    gat_final<<<(NN + 7) / 8, 256, 0, stream>>>(acc, x, bias, lnw, lnb, out);
}
